// DARTSCell_3169685865029
// MI455X (gfx1250) — compile-verified, matrix operand order checked by reading, not run on hardware
//
#include <hip/hip_runtime.h>
#include <hip/hip_bf16.h>

// ---------------------------------------------------------------------------
// DARTS RNN cell, MI455X (gfx1250).
//
// bf16 WMMA (v_wmma_f32_16x16x32_bf16, f32 accumulate); weights packed once
// per launch into WMMA B-fragment order (bf16, L2-resident ~40MB of 192MB);
// persistent cooperative kernel over the T=256 recurrence with a DAG-minimal
// global-barrier schedule; activations staged in LDS via the CDNA5 async
// path (global_load_async_to_lds_b128 + s_wait_asynccnt); rolled,
// software-pipelined inner loop (opaque trip count) so B fragments stream
// global->VGPR->WMMA without scratch spills.
// ---------------------------------------------------------------------------

typedef __attribute__((ext_vector_type(16))) __bf16 bf16x16;
typedef __attribute__((ext_vector_type(8)))  __bf16 bf16x8;
typedef __attribute__((ext_vector_type(4)))  __bf16 bf16x4;
typedef __attribute__((ext_vector_type(8)))  float  v8f;

#define NWG       16
#define NTHR      512
#define TSTEPS    256
#define BROWS     64
#define NHID      1024
#define LDS_ELEMS (BROWS * NHID)        // 65536 bf16 = 128 KB staged A matrix

// Hide a scalar from the optimizer (keeps loops rolled, loads in-loop).
__device__ __forceinline__ int opaque_s(int x) {
  asm volatile("" : "+s"(x));
  return x;
}

// Assumed gfx1250 16-bit WMMA operand K-mapping (mirrors ISA 7.12.2 A-layout):
// element e (0..15) of the per-lane v16bf fragment, lane half = lane>=16.
__device__ __host__ __forceinline__ int kOfElem(int lane, int e) {
  int v = e >> 1;
  return ((v & 4) << 2) | ((lane >> 4) << 3) | ((v & 3) << 1) | (e & 1);
}

// Genotype tables
__constant__ int g_PRED[8]  = {0, 1, 1, 1, 2, 5, 3, 5};
__constant__ int g_ACT[8]   = {0, 1, 1, 3, 2, 0, 2, 1};   // 0=sig 1=relu 2=tanh 3=id
__constant__ int g_WSLOT[8] = {1, 2, 3, -1, 5, -1, -1, -1}; // only states 0,1,2,3,5 re-read

// ---------------------------------------------------------------------------
// Prologue: convert W0/Ws f32 -> bf16 packed in WMMA B-fragment order,
// convert initial hidden state, reset the global barrier.
// Pack layout per matrix: frag(nt, kt) is 32 lanes x 16 bf16 contiguous
// (1 KB), frag index = nt*ktiles + kt  -> k-tiles stream contiguously.
// ---------------------------------------------------------------------------
__global__ void darts_pack(const float* __restrict__ W0,
                           const float* __restrict__ Ws,
                           const float* __restrict__ hidden0,
                           __bf16* __restrict__ packW0,
                           __bf16* __restrict__ packWs,
                           __bf16* __restrict__ hstate,
                           unsigned* __restrict__ bar)
{
  const long tid = (long)blockIdx.x * NTHR + threadIdx.x;
  const long FR_W0 = 128L * 64 * 32;        // 262144 fragment-lanes
  const long FR_WS = 8L * 128 * 32 * 32;    // 1048576
  const long HS4   = LDS_ELEMS / 4;         // 16384 float4 chunks

  if (tid < FR_W0) {
    const int lane = (int)(tid & 31);
    const int kt   = (int)((tid >> 5) & 63);
    const int nt   = (int)(tid >> 11);
    const int n    = nt * 16 + (lane & 15);
    __bf16* dst = packW0 + (((size_t)(nt * 64 + kt) * 32 + lane) * 16);
    #pragma unroll
    for (int e = 0; e < 16; ++e) {
      const int k = kt * 32 + kOfElem(lane, e);
      dst[e] = (__bf16)W0[(size_t)k * 2048 + n];
    }
  } else if (tid < FR_W0 + FR_WS) {
    const long r   = tid - FR_W0;
    const int lane = (int)(r & 31);
    const int kt   = (int)((r >> 5) & 31);
    const int nt   = (int)((r >> 10) & 127);
    const int m    = (int)(r >> 17);
    const int n    = nt * 16 + (lane & 15);
    const float* Wm = Ws + (size_t)m * 1024 * 2048;
    __bf16* dst = packWs + (size_t)m * (128 * 32 * 512)
                         + (((size_t)(nt * 32 + kt) * 32 + lane) * 16);
    #pragma unroll
    for (int e = 0; e < 16; ++e) {
      const int k = kt * 32 + kOfElem(lane, e);
      dst[e] = (__bf16)Wm[(size_t)k * 2048 + n];
    }
  } else if (tid < FR_W0 + FR_WS + HS4) {
    const long i = tid - FR_W0 - FR_WS;
    const float4 f = ((const float4*)hidden0)[i];
    bf16x4 b;
    b.x = (__bf16)f.x; b.y = (__bf16)f.y; b.z = (__bf16)f.z; b.w = (__bf16)f.w;
    ((bf16x4*)hstate)[i] = b;
  } else if (tid == FR_W0 + FR_WS + HS4) {
    bar[0] = 0u;  // arrive counter
    bar[1] = 0u;  // generation
  }
}

// ---------------------------------------------------------------------------
// Device helpers
// ---------------------------------------------------------------------------
__device__ __forceinline__ float sigmoidf_(float x) {
  return 1.f / (1.f + __expf(-x));
}
__device__ __forceinline__ float tanhf_(float x) {
  const float e = __expf(2.f * x);
  return 1.f - 2.f / (e + 1.f);
}
__device__ __forceinline__ float actf_(int a, float x) {
  switch (a) {
    case 0:  return sigmoidf_(x);
    case 1:  return fmaxf(x, 0.f);
    case 2:  return tanhf_(x);
    default: return x;
  }
}

__device__ __forceinline__ void gridBarrier(unsigned* bar) {
  __threadfence();
  __syncthreads();
  if (threadIdx.x == 0) {
    unsigned* cnt = bar;
    unsigned* gen = bar + 1;
    const unsigned g = __hip_atomic_load(gen, __ATOMIC_RELAXED, __HIP_MEMORY_SCOPE_AGENT);
    const unsigned a = __hip_atomic_fetch_add(cnt, 1u, __ATOMIC_ACQ_REL, __HIP_MEMORY_SCOPE_AGENT);
    if (a == (unsigned)(NWG - 1)) {
      __hip_atomic_store(cnt, 0u, __ATOMIC_RELAXED, __HIP_MEMORY_SCOPE_AGENT);
      __hip_atomic_fetch_add(gen, 1u, __ATOMIC_RELEASE, __HIP_MEMORY_SCOPE_AGENT);
    } else {
      while (__hip_atomic_load(gen, __ATOMIC_ACQUIRE, __HIP_MEMORY_SCOPE_AGENT) == g)
        __builtin_amdgcn_s_sleep(2);
    }
  }
  __syncthreads();
  __threadfence();
}

// Async LDS staging of 64x1024 bf16 (128 KB): CDNA5 ASYNCcnt path.
// GV form: LDS address from a VGPR; global addr in 64-bit VGPR pair.
// 512 threads x 16 b128 copies; each wave waits for its own ASYNCcnt before
// the workgroup barrier publishes the LDS tile.
__device__ __forceinline__ void ldsCopyAsync(__bf16* lds, const __bf16* src, int tid) {
  unsigned loff = (unsigned)(uintptr_t)lds + (unsigned)tid * 16u;
  unsigned long long gaddr =
      (unsigned long long)(uintptr_t)src + (unsigned long long)tid * 16ull;
  #pragma unroll
  for (int i = 0; i < (LDS_ELEMS * 2 / 16) / NTHR; ++i) {   // 16
    asm volatile("global_load_async_to_lds_b128 %0, %1, off"
                 :: "v"(loff), "v"(gaddr) : "memory");
    loff  += NTHR * 16u;
    gaddr += NTHR * 16ull;
  }
  asm volatile("s_wait_asynccnt 0x0" ::: "memory");
}

// f32 -> bf16 conversion staging for x_t (needs VALU, so classic path)
__device__ __forceinline__ void ldsConvF32(__bf16* lds, const float* src, int tid) {
  const float4* s = (const float4*)src;
  bf16x4* d = (bf16x4*)lds;
  #pragma unroll 4
  for (int i = 0; i < (LDS_ELEMS / 4) / NTHR; ++i) { // 32 iters
    const float4 f = s[tid + i * NTHR];
    bf16x4 b;
    b.x = (__bf16)f.x; b.y = (__bf16)f.y; b.z = (__bf16)f.z; b.w = (__bf16)f.w;
    d[tid + i * NTHR] = b;
  }
}

__device__ __forceinline__ v8f wmma_bf16(bf16x16 a, bf16x16 b, v8f c) {
  return __builtin_amdgcn_wmma_f32_16x16x32_bf16(
      /*neg_a=*/false, a, /*neg_b=*/false, b,
      /*c_mod=*/(short)0, c, /*reuse_a=*/false, /*reuse_b=*/false);
}

// A fragment: two contiguous 16B chunks of one LDS row, concatenated (SSA).
__device__ __forceinline__ bf16x16 loadA(const __bf16* ap) {
  const bf16x8 lo = *(const bf16x8*)ap;
  const bf16x8 hi = *(const bf16x8*)(ap + 16);
  return __builtin_shufflevector(lo, hi,
      0, 1, 2, 3, 4, 5, 6, 7, 8, 9, 10, 11, 12, 13, 14, 15);
}

// One wave accumulates a 16x16 c-tile and a 16x16 h-tile over [ktBeg,ktEnd).
// A from LDS; B streams from packed bf16 weights in L2. Trip count is
// laundered so the loop stays rolled (no full unroll -> no spill), with a
// depth-2 software pipeline on the B stream.
__device__ __forceinline__ void mm_tiles(v8f& accC, v8f& accH,
    const __bf16* __restrict__ lds, const __bf16* __restrict__ pack,
    int ktilesTotal, int ktBeg, int ktEnd, int ldsKtBase,
    int ntC, int rowBase, int lane)
{
  const int m     = lane & 15;
  const int khalf = (lane >> 4) << 3;
  const __bf16* aRow = lds + (size_t)(rowBase + m) * NHID + khalf
                           + (size_t)(ktBeg - ldsKtBase) * 32;
  const __bf16* bC = pack + ((size_t)ntC * ktilesTotal + ktBeg) * 512 + (size_t)lane * 16;
  const __bf16* bH = pack + ((size_t)(ntC + 64) * ktilesTotal + ktBeg) * 512 + (size_t)lane * 16;

  const int n = opaque_s(ktEnd - ktBeg);    // opaque trip count (== 32)

  bf16x16 fc = *(const bf16x16*)bC;
  bf16x16 fh = *(const bf16x16*)bH;
  #pragma unroll 2
  for (int k = 0; k < n - 1; ++k) {
    const bf16x16 a = loadA(aRow);
    const bf16x16 fc2 = *(const bf16x16*)(bC + 512);   // next-iteration B
    const bf16x16 fh2 = *(const bf16x16*)(bH + 512);
    __builtin_prefetch(bC + 4 * 512, 0, 1);            // global_prefetch ahead
    __builtin_prefetch(bH + 4 * 512, 0, 1);
    accC = wmma_bf16(a, fc, accC);
    accH = wmma_bf16(a, fh, accH);
    fc = fc2;
    fh = fh2;
    aRow += 32;
    bC += 512;
    bH += 512;
  }
  const bf16x16 a = loadA(aRow);
  accC = wmma_bf16(a, fc, accC);
  accH = wmma_bf16(a, fh, accH);
}

// ---------------------------------------------------------------------------
// Persistent recurrence kernel: 16 WGs x 512 threads (16 wave32 waves ->
// 4 waves per SIMD32 for latency hiding). WG owns 64 hidden columns; wave
// (mtile, colgroup) owns a 16x16 tile and computes both halves (c and h).
// ---------------------------------------------------------------------------
__global__ __launch_bounds__(NTHR, 1)
void darts_rnn(const float* __restrict__ x,
               const __bf16* __restrict__ packW0,
               const __bf16* __restrict__ packWs,
               __bf16* __restrict__ S,         // 6 state slots of 64x1024 bf16
               __bf16* __restrict__ hstate,    // h_prev, 64x1024 bf16
               unsigned* __restrict__ bar,
               float* __restrict__ out)
{
  extern __shared__ __align__(16) __bf16 ldsA[];   // 64x1024 bf16
  const int tid     = threadIdx.x;
  const int lane    = tid & 31;
  const int wave    = tid >> 5;                    // 0..15
  const int rowBase = (wave & 3) * 16;
  const int ntC     = (int)blockIdx.x * 4 + (wave >> 2);  // 0..63
  const int nloc    = lane & 15;
  const int rOff    = (lane >> 4) << 3;
  const int col     = ntC * 16 + nloc;

  for (int t = 0; t < TSTEPS; ++t) {
    // ---- step 0: [x_t, h_prev] @ W0, split K=2048 into two LDS phases ----
    v8f c0 = {}, h0 = {};
    __syncthreads();
    ldsConvF32(ldsA, x + (size_t)t * LDS_ELEMS, tid);        // A = x_t (bf16)
    __syncthreads();
    mm_tiles(c0, h0, ldsA, packW0, 64, 0, 32, 0, ntC, rowBase, lane);
    __syncthreads();
    ldsCopyAsync(ldsA, hstate, tid);                         // A = h_prev
    __syncthreads();
    mm_tiles(c0, h0, ldsA, packW0, 64, 32, 64, 32, ntC, rowBase, lane);
    // s0 = hp + sig(c0) * (tanh(h0) - hp); hp is resident in LDS
    #pragma unroll
    for (int i = 0; i < 8; ++i) {
      const int row = rowBase + rOff + i;
      const float sp = (float)ldsA[(size_t)row * NHID + col];
      const float sv = sp + sigmoidf_(c0[i]) * (tanhf_(h0[i]) - sp);
      S[(size_t)row * NHID + col] = (__bf16)sv;              // slot 0
    }
    gridBarrier(bar);

    // ---- genotype steps: s_{gi+1} = sp + sig(c)*(act(h)-sp), sp=S[pred] ----
    v8f acc = {};
    for (int gi = 0; gi < 8; ++gi) {
      __syncthreads();
      ldsCopyAsync(ldsA, S + (size_t)g_PRED[gi] * LDS_ELEMS, tid);
      __syncthreads();
      v8f cc = {}, hh = {};
      mm_tiles(cc, hh, ldsA, packWs + (size_t)gi * (128 * 32 * 512),
               32, 0, 32, 0, ntC, rowBase, lane);
      const int slot = g_WSLOT[gi];
      const int a    = g_ACT[gi];
      #pragma unroll
      for (int i = 0; i < 8; ++i) {
        const int row = rowBase + rOff + i;
        const float sp = (float)ldsA[(size_t)row * NHID + col];
        const float sv = sp + sigmoidf_(cc[i]) * (actf_(a, hh[i]) - sp);
        if (slot >= 0)
          S[(size_t)slot * LDS_ELEMS + (size_t)row * NHID + col] = (__bf16)sv;
        acc[i] += sv;
      }
      // DAG-minimal barrier schedule: S1 read at gi=1..3; S2,S3 read at
      // gi=4,6; S5 read at gi=5,7.
      if (gi == 0 || gi == 3 || gi == 4) gridBarrier(bar);
    }

    // ---- writeout: h_t = mean(s1..s8); NT stores keep weights hot in L2 ----
    #pragma unroll
    for (int i = 0; i < 8; ++i) {
      const int row = rowBase + rOff + i;
      const float hv = acc[i] * 0.125f;
      __builtin_nontemporal_store(hv, &out[(size_t)t * LDS_ELEMS + (size_t)row * NHID + col]);
      hstate[(size_t)row * NHID + col] = (__bf16)hv;
      if (t == TSTEPS - 1)
        __builtin_nontemporal_store(
            hv, &out[(size_t)TSTEPS * LDS_ELEMS + (size_t)row * NHID + col]);
    }
    gridBarrier(bar);
  }
}

// ---------------------------------------------------------------------------
// Host launcher
// ---------------------------------------------------------------------------
extern "C" void kernel_launch(void* const* d_in, const int* in_sizes, int n_in,
                              void* d_out, int out_size, void* d_ws, size_t ws_size,
                              hipStream_t stream)
{
  (void)in_sizes; (void)n_in; (void)out_size; (void)ws_size;
  const float* x      = (const float*)d_in[0];   // [256,64,1024]
  const float* hidden = (const float*)d_in[1];   // [1,64,1024]
  const float* W0     = (const float*)d_in[2];   // [2048,2048]
  const float* Ws     = (const float*)d_in[3];   // [8,1024,2048]
  float* out          = (float*)d_out;

  char* ws = (char*)d_ws;
  size_t off = 0;
  __bf16* packW0 = (__bf16*)(ws + off); off += (size_t)2048 * 2048 * 2;       // 8 MB
  __bf16* packWs = (__bf16*)(ws + off); off += (size_t)8 * 1024 * 2048 * 2;   // 32 MB
  __bf16* S      = (__bf16*)(ws + off); off += (size_t)6 * LDS_ELEMS * 2;     // 768 KB
  __bf16* hstate = (__bf16*)(ws + off); off += (size_t)LDS_ELEMS * 2;         // 128 KB
  unsigned* bar  = (unsigned*)(ws + off); off += 256;

  const long packThreads = 262144L + 1048576L + 16384L + 1L;
  const int  packBlocks  = (int)((packThreads + NTHR - 1) / NTHR);
  darts_pack<<<packBlocks, NTHR, 0, stream>>>(W0, Ws, hidden, packW0, packWs, hstate, bar);

  darts_rnn<<<NWG, NTHR, LDS_ELEMS * sizeof(__bf16), stream>>>(
      x, packW0, packWs, S, hstate, bar, out);
}
